// preGATConvPrior_4784593568252
// MI455X (gfx1250) — compile-verified, matrix operand order checked by reading, not run on hardware
//
#include <hip/hip_runtime.h>

// ---------------- problem constants (from reference) ----------------
#define FIN 256   // in_feats
#define NH  4     // heads
#define HD  128   // H*D (out feats total)

typedef __attribute__((ext_vector_type(16))) _Float16 v16h;
typedef __attribute__((ext_vector_type(8)))  float    v8f;

__device__ __forceinline__ void cvt4(v16h& v, int base, float4 f) {
  v[base + 0] = (_Float16)f.x;
  v[base + 1] = (_Float16)f.y;
  v[base + 2] = (_Float16)f.z;
  v[base + 3] = (_Float16)f.w;
}

// ---------------------------------------------------------------
// Kernel 1: ft = feat @ w_fc^T  via v_wmma_f32_16x16x32_f16
// grid = n/16 blocks (one 16-row M tile per block),
// 8 waves per block = the 8 column tiles of the 128-wide output.
// ---------------------------------------------------------------
__global__ __launch_bounds__(256) void gat_proj_wmma(
    const float* __restrict__ feat, const float* __restrict__ w_fc,
    float* __restrict__ ft)
{
  const int lane  = threadIdx.x & 31;
  const int wave  = threadIdx.x >> 5;
  const int mtile = blockIdx.x * 16;   // output row base
  const int jtile = wave * 16;         // output col base (0..112)
  const int half  = lane >> 4;         // 0: k-halves {0..7,16..23}; 1: {8..15,24..31}
  const int l     = lane & 15;

  // A: one feat row per lane.  B: rows of w_fc are rows of B^T -> same layout.
  const float* arow = feat + (size_t)(mtile + l) * FIN + half * 8;
  const float* brow = w_fc + (size_t)(jtile + l) * FIN + half * 8;

  v8f c = {};
#pragma unroll
  for (int kt = 0; kt < FIN; kt += 32) {
    const float4* ap = (const float4*)(arow + kt);
    const float4* bp = (const float4*)(brow + kt);
    v16h a, b;
    cvt4(a, 0, ap[0]); cvt4(a, 4, ap[1]);   // k = base+0..7
    cvt4(a, 8, ap[4]); cvt4(a, 12, ap[5]);  // k = base+16..23
    cvt4(b, 0, bp[0]); cvt4(b, 4, bp[1]);
    cvt4(b, 8, bp[4]); cvt4(b, 12, bp[5]);
    // (neg_a, A, neg_b, B, c_mod, C, reuse_a, reuse_b)
    c = __builtin_amdgcn_wmma_f32_16x16x32_f16(false, a, false, b,
                                               (short)0, c, false, false);
  }

  // D layout: VGPR r holds row (r + 8*half), col = l
  float* drow = ft + (size_t)(mtile + half * 8) * HD + jtile + l;
#pragma unroll
  for (int r = 0; r < 8; ++r) drow[(size_t)r * HD] = c[r];
}

// ---------------------------------------------------------------
// Kernel 2: init out = bias (broadcast), m = -inf, s = 0
// ---------------------------------------------------------------
__global__ void gat_init(float* __restrict__ out, const float* __restrict__ bias,
                         float* __restrict__ m, float* __restrict__ s, int n)
{
  int gid = blockIdx.x * 256 + threadIdx.x;
  if (gid < n * HD) out[gid] = bias[gid & (HD - 1)];
  if (gid < n * NH) {
    ((unsigned int*)m)[gid] = 0xFF800000u;  // -inf
    s[gid] = 0.0f;
  }
}

// ---------------------------------------------------------------
// Kernel 3: segment max of e_w over dst (per head), via int-bit atomics
// ---------------------------------------------------------------
__global__ void gat_segmax(const float* __restrict__ e_w,
                           const int* __restrict__ dst,
                           float* __restrict__ m, int e)
{
  int gid = blockIdx.x * 256 + threadIdx.x;
  if (gid >= e * NH) return;
  int edge = gid >> 2, h = gid & 3;
  float v = e_w[gid];
  int* addr = (int*)(m + (size_t)dst[edge] * NH + h);
  if (v >= 0.0f) atomicMax(addr, __float_as_int(v));
  else           atomicMin((unsigned int*)addr, __float_as_uint(v));
}

// ---------------------------------------------------------------
// Kernel 4: s[dst,h] += exp(e_w - m[dst,h])
// ---------------------------------------------------------------
__global__ void gat_segsum(const float* __restrict__ e_w,
                           const int* __restrict__ dst,
                           const float* __restrict__ m,
                           float* __restrict__ s, int e)
{
  int gid = blockIdx.x * 256 + threadIdx.x;
  if (gid >= e * NH) return;
  int edge = gid >> 2, h = gid & 3;
  int d0 = dst[edge];
  float mm = m[(size_t)d0 * NH + h];
  if (!(mm > -3.0e38f && mm < 3.0e38f)) mm = 0.0f;   // isfinite guard
  atomicAdd(s + (size_t)d0 * NH + h, __expf(e_w[gid] - mm));
}

// ---------------------------------------------------------------
// Kernel 5: normalized attention a = exp(e_w - m[dst]) / s[dst]
// ---------------------------------------------------------------
__global__ void gat_norm(const float* __restrict__ e_w,
                         const int* __restrict__ dst,
                         const float* __restrict__ m,
                         const float* __restrict__ s,
                         float* __restrict__ a, int e)
{
  int gid = blockIdx.x * 256 + threadIdx.x;
  if (gid >= e * NH) return;
  int edge = gid >> 2, h = gid & 3;
  int d0 = dst[edge];
  float mm = m[(size_t)d0 * NH + h];
  if (!(mm > -3.0e38f && mm < 3.0e38f)) mm = 0.0f;
  a[gid] = __expf(e_w[gid] - mm) / s[(size_t)d0 * NH + h];
}

// ---------------------------------------------------------------
// Kernel 6: out[dst] += ft[src] * a   (thread per (edge, feature))
// One wave = 32 consecutive features of one edge -> coalesced 128B
// gather of ft[src] (L2-resident) + coalesced f32 atomic adds.
// ---------------------------------------------------------------
__global__ void gat_agg(const float* __restrict__ a,
                        const int* __restrict__ src,
                        const int* __restrict__ dst,
                        const float* __restrict__ ft,
                        float* __restrict__ out, int e)
{
  unsigned int gid = blockIdx.x * 256u + threadIdx.x;
  if (gid >= (unsigned int)e * HD) return;
  unsigned int edge = gid >> 7;
  unsigned int f    = gid & (HD - 1);
  unsigned int h    = f >> 5;
  float av = a[(size_t)edge * NH + h];
  float v  = ft[(size_t)src[edge] * HD + f] * av;
  atomicAdd(out + (size_t)dst[edge] * HD + f, v);
}

// ---------------------------------------------------------------
extern "C" void kernel_launch(void* const* d_in, const int* in_sizes, int n_in,
                              void* d_out, int out_size, void* d_ws, size_t ws_size,
                              hipStream_t stream)
{
  const float* feat = (const float*)d_in[0];   // [N, 256]
  const float* w_fc = (const float*)d_in[1];   // [128, 256]
  const float* bias = (const float*)d_in[2];   // [128]
  const float* e_w  = (const float*)d_in[3];   // [E, 4, 1]
  const int*   src  = (const int*)d_in[4];     // [E]
  const int*   dst  = (const int*)d_in[5];     // [E]
  float* out = (float*)d_out;                  // [N, 4, 32]

  const int n = in_sizes[0] / FIN;             // 100000
  const int e = in_sizes[4];                   // 1600000

  // workspace layout
  float* ft = (float*)d_ws;                    // n*128
  float* m  = ft + (size_t)n * HD;             // n*4
  float* s  = m  + (size_t)n * NH;             // n*4
  float* a  = s  + (size_t)n * NH;             // e*4

  int t;

  t = n * HD;
  gat_init<<<(t + 255) / 256, 256, 0, stream>>>(out, bias, m, s, n);

  gat_proj_wmma<<<n / 16, 256, 0, stream>>>(feat, w_fc, ft);

  t = e * NH;
  gat_segmax<<<(t + 255) / 256, 256, 0, stream>>>(e_w, dst, m, e);
  gat_segsum<<<(t + 255) / 256, 256, 0, stream>>>(e_w, dst, m, s, e);
  gat_norm  <<<(t + 255) / 256, 256, 0, stream>>>(e_w, dst, m, s, a, e);

  unsigned long long tt = (unsigned long long)e * HD;
  gat_agg<<<(unsigned int)((tt + 255) / 256), 256, 0, stream>>>(a, src, dst, ft, out, e);
}
